// MultiScaleAttentionBlock_3770981286358
// MI455X (gfx1250) — matrix operand order repaired and checked by reading, not compiled, not run on hardware
//
#include <hip/hip_runtime.h>
#include <hip/hip_bf16.h>
#include <math.h>
#include <stdint.h>

typedef __bf16 bf16_t;
typedef __attribute__((ext_vector_type(16))) __bf16 v16bf;
typedef __attribute__((ext_vector_type(8)))  __bf16 v8bf;
typedef __attribute__((ext_vector_type(8)))  float  v8f;

#define DEV __device__ __forceinline__

// ---------------------------------------------------------------------------
// WMMA helpers (CDNA5 gfx1250, wave32). D = A(16x32 bf16) * B(32x16 bf16) + C
// ---------------------------------------------------------------------------
DEV v8f wmma_bf16(v16bf a, v16bf b, v8f c) {
  return __builtin_amdgcn_wmma_f32_16x16x32_bf16(
      /*neg_a=*/false, a, /*neg_b=*/false, b,
      /*c_mod=*/(short)0, c, /*reuse_a=*/false, /*reuse_b=*/false);
}

// A-matrix 16x32 bf16 fragment (ISA 7.12.2): row M = lane&15, half = lane>>4;
// K = half*16 + i (contiguous, matches load_bT_frag).
DEV v16bf load_a_frag(const bf16_t* __restrict__ base, int ld, int row0, int k0, int lane) {
  int r   = row0 + (lane & 15);
  int hlf = lane >> 4;
  const bf16_t* p = base + (size_t)r * ld + k0 + hlf * 16;
  v8bf lo = *(const v8bf*)(p);
  v8bf hi = *(const v8bf*)(p + 8);
  v16bf out;
#pragma unroll
  for (int i = 0; i < 8; ++i) { out[i] = lo[i]; out[i + 8] = hi[i]; }
  return out;
}

// B-matrix 32x16 bf16 fragment from a row-major matrix whose ROW index is the
// WMMA N dim and COLUMN index is K: lane N = lane&15, K = (lane>>4)*16 + i.
DEV v16bf load_bT_frag(const bf16_t* __restrict__ base, int ld, int row0, int k0, int lane) {
  int r   = row0 + (lane & 15);
  int hlf = lane >> 4;
  const bf16_t* p = base + (size_t)r * ld + k0 + hlf * 16;
  v8bf lo = *(const v8bf*)(p);
  v8bf hi = *(const v8bf*)(p + 8);
  v16bf out;
#pragma unroll
  for (int i = 0; i < 8; ++i) { out[i] = lo[i]; out[i + 8] = hi[i]; }
  return out;
}

// Packed-bf16 helpers for the softmax row-max butterfly (exactness of online
// softmax only needs a consistent shift, so bf16-precision max is valid).
DEV unsigned pack2_bf16(float lo, float hi) {
  unsigned short l = __builtin_bit_cast(unsigned short, (bf16_t)lo);
  unsigned short h = __builtin_bit_cast(unsigned short, (bf16_t)hi);
  return (unsigned)l | ((unsigned)h << 16);
}
DEV float bf16lo_f32(unsigned u) { return __builtin_bit_cast(float, u << 16); }
DEV float bf16hi_f32(unsigned u) { return __builtin_bit_cast(float, u & 0xffff0000u); }
DEV unsigned pk_max_bf16(unsigned a, unsigned b) {
  unsigned d;
  asm("v_pk_max_num_bf16 %0, %1, %2" : "=v"(d) : "v"(a), "v"(b));
  return d;
}

// ---------------------------------------------------------------------------
// GEMM: C[M,N] = A[M,K](bf16) * W[N,K]^T(bf16) + bias[N](f32)
// block = 256 threads = 8 waves, block tile 64(M) x 128(N), wave tile 32x32.
// ---------------------------------------------------------------------------
enum { EPI_F32 = 0, EPI_BF16 = 1, EPI_GELU_BF16 = 2 };

template <int EPI>
__global__ __launch_bounds__(256)
void gemm_bf16_kernel(const bf16_t* __restrict__ A, const bf16_t* __restrict__ W,
                      const float* __restrict__ bias, float* __restrict__ Cf,
                      bf16_t* __restrict__ Cb, int M, int N, int K) {
  const int lane = threadIdx.x & 31;
  const int wave = threadIdx.x >> 5;
  const int m0 = blockIdx.x * 64 + (wave >> 2) * 32;
  const int n0 = blockIdx.y * 128 + (wave & 3) * 32;
  v8f acc00 = {}, acc01 = {}, acc10 = {}, acc11 = {};
  for (int k0 = 0; k0 < K; k0 += 32) {
    if (k0 + 32 < K) {
      __builtin_prefetch(A + (size_t)(m0 + (lane & 15)) * K + k0 + 32, 0, 0);
      __builtin_prefetch(W + (size_t)(n0 + (lane & 15)) * K + k0 + 32, 0, 0);
    }
    v16bf a0 = load_a_frag(A, K, m0,      k0, lane);
    v16bf a1 = load_a_frag(A, K, m0 + 16, k0, lane);
    v16bf b0 = load_bT_frag(W, K, n0,      k0, lane);
    v16bf b1 = load_bT_frag(W, K, n0 + 16, k0, lane);
    acc00 = wmma_bf16(a0, b0, acc00);
    acc01 = wmma_bf16(a0, b1, acc01);
    acc10 = wmma_bf16(a1, b0, acc10);
    acc11 = wmma_bf16(a1, b1, acc11);
  }
  const int col = lane & 15, hlf = lane >> 4;
#pragma unroll
  for (int tm = 0; tm < 2; ++tm) {
#pragma unroll
    for (int tn = 0; tn < 2; ++tn) {
      v8f acc = (tm == 0) ? ((tn == 0) ? acc00 : acc01) : ((tn == 0) ? acc10 : acc11);
      int n = n0 + tn * 16 + col;
      float bv = bias[n];
#pragma unroll
      for (int r = 0; r < 8; ++r) {
        int m = m0 + tm * 16 + r + 8 * hlf;
        float v = acc[r] + bv;
        if (EPI == EPI_GELU_BF16) v = 0.5f * v * (1.0f + erff(v * 0.70710678118654752f));
        if (EPI == EPI_F32) Cf[(size_t)m * N + n] = v;
        else                Cb[(size_t)m * N + n] = (bf16_t)v;
      }
    }
  }
}

// ---------------------------------------------------------------------------
// Flash attention over packed qkv [B*S, 1536] (q|k|v each D=512, head = 64).
// block = 256 threads = 8 waves; wave owns 16 q rows; block owns 128 q rows
// of one (b,h). K tile staged via async global->LDS; V staged transposed.
// Row sums computed with a ones-matrix WMMA; row max via packed-bf16 butterfly.
// ---------------------------------------------------------------------------
__global__ __launch_bounds__(256)
void attn_kernel(const bf16_t* __restrict__ qkv, bf16_t* __restrict__ O, int Sseq) {
  __shared__ bf16_t Kt[32 * 64];        // [key][hd]
  __shared__ bf16_t Vt[64 * 32];        // [hd][key]  (transposed)
  __shared__ bf16_t Pb[8][16 * 32];     // per-wave P staging [qrow][key]
  const int lane = threadIdx.x & 31;
  const int wave = threadIdx.x >> 5;
  const int b = blockIdx.z, h = blockIdx.y;
  const int q0 = blockIdx.x * 128 + wave * 16;
  const size_t rowbase = (size_t)b * Sseq * 1536;
  const bf16_t* qb  = qkv + rowbase + h * 64;
  const bf16_t* kbp = qkv + rowbase + 512 + h * 64;
  const bf16_t* vbp = qkv + rowbase + 1024 + h * 64;

  v16bf qa0 = load_a_frag(qb, 1536, q0, 0,  lane);
  v16bf qa1 = load_a_frag(qb, 1536, q0, 32, lane);
  v16bf ones;
#pragma unroll
  for (int i = 0; i < 16; ++i) ones[i] = (bf16_t)1.0f;

  float mrun[8], lrun[8];
  v8f o0 = {}, o1 = {}, o2 = {}, o3 = {};
#pragma unroll
  for (int r = 0; r < 8; ++r) { mrun[r] = -1e30f; lrun[r] = 0.0f; }

  const int col = lane & 15, hlf = lane >> 4;
  const int row = threadIdx.x >> 3;        // 0..31 (coop staging)
  const int c8  = (threadIdx.x & 7) * 8;   // hd column start
  const unsigned ldsK = (unsigned)(uintptr_t)(&Kt[row * 64 + c8]);

  for (int kblk = 0; kblk < Sseq; kblk += 32) {
    __syncthreads();
    {
      // K tile: async copy straight into LDS (ASYNCcnt-tracked, no VGPR return)
      unsigned long long gk =
          (unsigned long long)(uintptr_t)(kbp + (size_t)(kblk + row) * 1536 + c8);
      asm volatile("global_load_async_to_lds_b128 %0, %1, off"
                   :: "v"(ldsK), "v"(gk) : "memory");
      // V tile: through VGPRs (needs transpose into Vt[hd][key])
      v8bf vd = *(const v8bf*)(vbp + (size_t)(kblk + row) * 1536 + c8);
#pragma unroll
      for (int i = 0; i < 8; ++i) Vt[(c8 + i) * 32 + row] = vd[i];
      asm volatile("s_wait_asynccnt 0x0" ::: "memory");
    }
    __syncthreads();

    // scores: S[16q x 32k] as two 16x16 f32 tiles
    v8f s0 = {}, s1 = {};
    {
      v16bf kb0 = load_bT_frag(Kt, 64, 0,  0,  lane);
      v16bf kb1 = load_bT_frag(Kt, 64, 0,  32, lane);
      s0 = wmma_bf16(qa1, kb1, wmma_bf16(qa0, kb0, s0));
      v16bf kc0 = load_bT_frag(Kt, 64, 16, 0,  lane);
      v16bf kc1 = load_bT_frag(Kt, 64, 16, 32, lane);
      s1 = wmma_bf16(qa1, kc1, wmma_bf16(qa0, kc0, s1));
    }
    const float sc = 0.125f;  // 1/sqrt(64)
    // per-lane row maxima, packed pairs of rows as bf16
    unsigned pk[4];
#pragma unroll
    for (int r2 = 0; r2 < 4; ++r2) {
      float e0 = s0[2 * r2] * sc,     f0 = s1[2 * r2] * sc;
      float e1 = s0[2 * r2 + 1] * sc, f1 = s1[2 * r2 + 1] * sc;
      s0[2 * r2] = e0; s1[2 * r2] = f0;
      s0[2 * r2 + 1] = e1; s1[2 * r2 + 1] = f1;
      pk[r2] = pack2_bf16(fmaxf(e0, f0), fmaxf(e1, f1));
    }
#pragma unroll
    for (int mask = 1; mask <= 8; mask <<= 1) {
#pragma unroll
      for (int r2 = 0; r2 < 4; ++r2) {
        unsigned other = (unsigned)__shfl_xor((int)pk[r2], mask, 32);
        pk[r2] = pk_max_bf16(pk[r2], other);
      }
    }
    float nm[8], al[8];
#pragma unroll
    for (int r2 = 0; r2 < 4; ++r2) {
      float bm0 = bf16lo_f32(pk[r2]), bm1 = bf16hi_f32(pk[r2]);
      nm[2 * r2]     = fmaxf(mrun[2 * r2], bm0);
      nm[2 * r2 + 1] = fmaxf(mrun[2 * r2 + 1], bm1);
    }
#pragma unroll
    for (int r = 0; r < 8; ++r) {
      al[r] = __expf(mrun[r] - nm[r]);
      mrun[r] = nm[r];
      float p0 = __expf(s0[r] - nm[r]);
      float p1 = __expf(s1[r] - nm[r]);
      Pb[wave][(r + 8 * hlf) * 32 + col]      = (bf16_t)p0;
      Pb[wave][(r + 8 * hlf) * 32 + 16 + col] = (bf16_t)p1;
      o0[r] = o0[r] * al[r]; o1[r] = o1[r] * al[r];
      o2[r] = o2[r] * al[r]; o3[r] = o3[r] * al[r];
    }
    // P as A-fragment (16x32); row sums via ones-matrix WMMA (replicated cols)
    v16bf pa = load_a_frag(&Pb[wave][0], 32, 0, 0, lane);
    v8f rs = {};
    rs = wmma_bf16(pa, ones, rs);
#pragma unroll
    for (int r = 0; r < 8; ++r) lrun[r] = lrun[r] * al[r] + rs[r];

    v16bf vb0 = load_bT_frag(Vt, 32, 0,  0, lane);
    v16bf vb1 = load_bT_frag(Vt, 32, 16, 0, lane);
    v16bf vb2 = load_bT_frag(Vt, 32, 32, 0, lane);
    v16bf vb3 = load_bT_frag(Vt, 32, 48, 0, lane);
    o0 = wmma_bf16(pa, vb0, o0);
    o1 = wmma_bf16(pa, vb1, o1);
    o2 = wmma_bf16(pa, vb2, o2);
    o3 = wmma_bf16(pa, vb3, o3);
  }
#pragma unroll
  for (int r = 0; r < 8; ++r) {
    size_t m = (size_t)b * Sseq + q0 + r + 8 * hlf;
    float inv = 1.0f / lrun[r];
    bf16_t* orow = O + m * 512 + h * 64 + col;
    orow[0]  = (bf16_t)(o0[r] * inv);
    orow[16] = (bf16_t)(o1[r] * inv);
    orow[32] = (bf16_t)(o2[r] * inv);
    orow[48] = (bf16_t)(o3[r] * inv);
  }
}

// ---------------------------------------------------------------------------
// LayerNorm(base + lerp(add)) * g + beta. One 256-thread block per row (D=512).
// S_in==S_out degrades the interp to an identity residual add.
// ---------------------------------------------------------------------------
template <bool WRITE_BF16>
__global__ __launch_bounds__(256)
void ln_kernel(const float* __restrict__ base, const float* __restrict__ add,
               const float* __restrict__ g, const float* __restrict__ beta,
               float* __restrict__ outf, bf16_t* __restrict__ outb,
               int S_out, int S_in) {
  __shared__ float red[256];
  const int row = blockIdx.x;
  const int b = row / S_out, s = row - b * S_out;
  float scale = (float)S_in / (float)S_out;
  float src = (s + 0.5f) * scale - 0.5f;
  src = fminf(fmaxf(src, 0.0f), (float)(S_in - 1));
  int i0 = (int)floorf(src);
  int i1 = (i0 + 1 < S_in) ? i0 + 1 : S_in - 1;
  float w = src - (float)i0;
  const float* a0 = add + ((size_t)b * S_in + i0) * 512;
  const float* a1 = add + ((size_t)b * S_in + i1) * 512;
  const float* xr = base + (size_t)row * 512;
  const int t = threadIdx.x;
  float v0 = xr[t]       + (1.0f - w) * a0[t]       + w * a1[t];
  float v1 = xr[t + 256] + (1.0f - w) * a0[t + 256] + w * a1[t + 256];
  red[t] = v0 + v1;
  __syncthreads();
  for (int st = 128; st > 0; st >>= 1) { if (t < st) red[t] += red[t + st]; __syncthreads(); }
  float mean = red[0] * (1.0f / 512.0f);
  __syncthreads();
  float d0 = v0 - mean, d1 = v1 - mean;
  red[t] = d0 * d0 + d1 * d1;
  __syncthreads();
  for (int st = 128; st > 0; st >>= 1) { if (t < st) red[t] += red[t + st]; __syncthreads(); }
  float rstd = rsqrtf(red[0] * (1.0f / 512.0f) + 1e-5f);
  float y0 = d0 * rstd * g[t] + beta[t];
  float y1 = d1 * rstd * g[t + 256] + beta[t + 256];
  if (outf) { outf[(size_t)row * 512 + t] = y0; outf[(size_t)row * 512 + t + 256] = y1; }
  if (WRITE_BF16) {
    outb[(size_t)row * 512 + t]       = (bf16_t)y0;
    outb[(size_t)row * 512 + t + 256] = (bf16_t)y1;
  }
}

__global__ void cvt_f32_bf16_kernel(const float* __restrict__ in, bf16_t* __restrict__ out, int n) {
  int i = blockIdx.x * blockDim.x + threadIdx.x;
  if (i < n) out[i] = (bf16_t)in[i];
}

// avg_pool1d(k=2,s=2) over S, fp32 -> bf16
__global__ void pool_kernel(const float* __restrict__ x, bf16_t* __restrict__ out, int n) {
  int i = blockIdx.x * blockDim.x + threadIdx.x;
  if (i >= n) return;
  int d = i & 511;
  int p = i >> 9;            // pooled row (global)
  int b = p >> 9;            // / 512
  int sp = p & 511;
  size_t src = ((size_t)b * 1024 + 2 * sp) * 512 + d;
  out[i] = (bf16_t)(0.5f * (x[src] + x[src + 512]));
}

// ---------------------------------------------------------------------------
extern "C" void kernel_launch(void* const* d_in, const int* in_sizes, int n_in,
                              void* d_out, int out_size, void* d_ws, size_t ws_size,
                              hipStream_t stream) {
  const float* x      = (const float*)d_in[0];
  const float* w_in1  = (const float*)d_in[1];
  const float* b_in1  = (const float*)d_in[2];
  const float* w_out1 = (const float*)d_in[3];
  const float* b_out1 = (const float*)d_in[4];
  const float* w_in2  = (const float*)d_in[5];
  const float* b_in2  = (const float*)d_in[6];
  const float* w_out2 = (const float*)d_in[7];
  const float* b_out2 = (const float*)d_in[8];
  const float* g1  = (const float*)d_in[9];
  const float* be1 = (const float*)d_in[10];
  const float* g2  = (const float*)d_in[11];
  const float* be2 = (const float*)d_in[12];
  const float* g3  = (const float*)d_in[13];
  const float* be3 = (const float*)d_in[14];
  const float* w_ff1 = (const float*)d_in[15];
  const float* b_ff1 = (const float*)d_in[16];
  const float* w_ff2 = (const float*)d_in[17];
  const float* b_ff2 = (const float*)d_in[18];

  constexpr int M1 = 8 * 1024;   // B*S
  constexpr int M2 = 8 * 512;    // B*S/2

  char* ws = (char*)d_ws;
  size_t off = 0;
  auto alloc = [&](size_t bytes) { char* p = ws + off; off += (bytes + 255) & ~(size_t)255; return p; };

  // persistent
  bf16_t* wb_in1  = (bf16_t*)alloc((size_t)1536 * 512 * 2);
  bf16_t* wb_out1 = (bf16_t*)alloc((size_t)512 * 512 * 2);
  bf16_t* wb_in2  = (bf16_t*)alloc((size_t)1536 * 512 * 2);
  bf16_t* wb_out2 = (bf16_t*)alloc((size_t)512 * 512 * 2);
  bf16_t* wb_ff1  = (bf16_t*)alloc((size_t)2048 * 512 * 2);
  bf16_t* wb_ff2  = (bf16_t*)alloc((size_t)512 * 2048 * 2);
  float*  x1f     = (float*)alloc((size_t)M1 * 512 * 4);
  float*  x2f     = (float*)alloc((size_t)M1 * 512 * 4);
  bf16_t* x2b     = (bf16_t*)alloc((size_t)M1 * 512 * 2);
  char* scr = ws + off;

  // phase-overlaid scratch (stream order guarantees safe reuse)
  size_t so = 0;
  auto salloc = [&](size_t bytes) { char* p = scr + so; so += (bytes + 255) & ~(size_t)255; return p; };
  // phase 1
  bf16_t* xb   = (bf16_t*)salloc((size_t)M1 * 512 * 2);
  bf16_t* qkv1 = (bf16_t*)salloc((size_t)M1 * 1536 * 2);
  bf16_t* O1   = (bf16_t*)salloc((size_t)M1 * 512 * 2);
  float*  a1   = (float*)salloc((size_t)M1 * 512 * 4);
  // phase 2 (overlays phase 1; a1 consumed by ln1 before these are written)
  so = 0;
  bf16_t* poolb = (bf16_t*)salloc((size_t)M2 * 512 * 2);
  bf16_t* qkv2  = (bf16_t*)salloc((size_t)M2 * 1536 * 2);
  bf16_t* O2    = (bf16_t*)salloc((size_t)M2 * 512 * 2);
  float*  a2    = (float*)salloc((size_t)M2 * 512 * 4);
  // phase 3 (overlays phase 2; a2 consumed by ln2 first)
  so = 0;
  bf16_t* hbuf = (bf16_t*)salloc((size_t)M1 * 2048 * 2);
  float*  ffb  = (float*)salloc((size_t)M1 * 512 * 4);

  auto cdiv = [](int a, int b) { return (a + b - 1) / b; };

  // bf16 mirrors of x and all GEMM weights
  cvt_f32_bf16_kernel<<<cdiv(M1 * 512, 256), 256, 0, stream>>>(x, xb, M1 * 512);
  cvt_f32_bf16_kernel<<<cdiv(1536 * 512, 256), 256, 0, stream>>>(w_in1, wb_in1, 1536 * 512);
  cvt_f32_bf16_kernel<<<cdiv(512 * 512, 256), 256, 0, stream>>>(w_out1, wb_out1, 512 * 512);
  cvt_f32_bf16_kernel<<<cdiv(1536 * 512, 256), 256, 0, stream>>>(w_in2, wb_in2, 1536 * 512);
  cvt_f32_bf16_kernel<<<cdiv(512 * 512, 256), 256, 0, stream>>>(w_out2, wb_out2, 512 * 512);
  cvt_f32_bf16_kernel<<<cdiv(2048 * 512, 256), 256, 0, stream>>>(w_ff1, wb_ff1, 2048 * 512);
  cvt_f32_bf16_kernel<<<cdiv(512 * 2048, 256), 256, 0, stream>>>(w_ff2, wb_ff2, 512 * 2048);

  // ---- MHA 1 ----
  gemm_bf16_kernel<EPI_BF16><<<dim3(M1 / 64, 1536 / 128), 256, 0, stream>>>(
      xb, wb_in1, b_in1, nullptr, qkv1, M1, 1536, 512);
  attn_kernel<<<dim3(1024 / 128, 8, 8), 256, 0, stream>>>(qkv1, O1, 1024);
  gemm_bf16_kernel<EPI_F32><<<dim3(M1 / 64, 512 / 128), 256, 0, stream>>>(
      O1, wb_out1, b_out1, a1, nullptr, M1, 512, 512);
  ln_kernel<false><<<M1, 256, 0, stream>>>(x, a1, g1, be1, x1f, nullptr, 1024, 1024);

  // ---- pooled MHA 2 + linear upsample ----
  pool_kernel<<<cdiv(M2 * 512, 256), 256, 0, stream>>>(x1f, poolb, M2 * 512);
  gemm_bf16_kernel<EPI_BF16><<<dim3(M2 / 64, 1536 / 128), 256, 0, stream>>>(
      poolb, wb_in2, b_in2, nullptr, qkv2, M2, 1536, 512);
  attn_kernel<<<dim3(512 / 128, 8, 8), 256, 0, stream>>>(qkv2, O2, 512);
  gemm_bf16_kernel<EPI_F32><<<dim3(M2 / 64, 512 / 128), 256, 0, stream>>>(
      O2, wb_out2, b_out2, a2, nullptr, M2, 512, 512);
  ln_kernel<true><<<M1, 256, 0, stream>>>(x1f, a2, g2, be2, x2f, x2b, 1024, 512);

  // ---- FFN ----
  gemm_bf16_kernel<EPI_GELU_BF16><<<dim3(M1 / 64, 2048 / 128), 256, 0, stream>>>(
      x2b, wb_ff1, b_ff1, nullptr, hbuf, M1, 2048, 512);
  gemm_bf16_kernel<EPI_F32><<<dim3(M1 / 64, 512 / 128), 256, 0, stream>>>(
      hbuf, wb_ff2, b_ff2, ffb, nullptr, M1, 512, 2048);
  ln_kernel<false><<<M1, 256, 0, stream>>>(x2f, ffb, g3, be3, (float*)d_out, nullptr, 1024, 1024);
}
